// MeanShift_88553635709560
// MI455X (gfx1250) — matrix operand order repaired and checked by reading, not compiled, not run on hardware
//
#include <hip/hip_runtime.h>

// MeanShift loss, fused streaming implementation for gfx1250 (MI455X).
// Memory floor: one pass over the 262MB unsup bank (~11us @ 23.3 TB/s).
// Compute: v_wmma_f32_16x16x32_bf16 (f32 accumulate).
// Bank tiles staged into LDS via the Tensor Data Mover (TENSOR_LOAD_TO_LDS,
// TENSORcnt-tracked) when the builtin is available; VGPR staging otherwise.

#define B_ROWS     256
#define DIMK       512
#define GROUP_COLS 512
#define SUB_COLS   16
#define NSUB       (GROUP_COLS / SUB_COLS)
#define KTOP       5

#if defined(__has_builtin)
#if __has_builtin(__builtin_amdgcn_tensor_load_to_lds) && \
    __has_builtin(__builtin_amdgcn_s_wait_tensorcnt)
#define MS_USE_TDM 1
#endif
#endif

typedef __attribute__((ext_vector_type(16))) __bf16 v16bf;
typedef __attribute__((ext_vector_type(8)))  float  v8f;
#ifdef MS_USE_TDM
typedef __attribute__((ext_vector_type(4))) unsigned v4u;
typedef __attribute__((ext_vector_type(4))) int      v4i;
typedef __attribute__((ext_vector_type(8))) int      v8i;
#endif

union FragBF { v16bf v; unsigned u[8]; };

__device__ __forceinline__ v8f zero8() {
  v8f z = {0.f, 0.f, 0.f, 0.f, 0.f, 0.f, 0.f, 0.f};
  return z;
}

__device__ __forceinline__ unsigned bf16_rne(float f) {
  unsigned u = __builtin_bit_cast(unsigned, f);
  unsigned r = 0x7FFFu + ((u >> 16) & 1u);
  return (u + r) >> 16;
}
__device__ __forceinline__ unsigned pack2_bf16(float lo, float hi) {
  return bf16_rne(lo) | (bf16_rne(hi) << 16);
}

// insert (dk,dq) into sorted-ascending-by-dk top-5 register list
__device__ __forceinline__ void ins5(float dk, float dq,
                                     float (&K)[KTOP], float (&Q)[KTOP]) {
  if (dk < K[KTOP - 1]) {
    K[KTOP - 1] = dk; Q[KTOP - 1] = dq;
#pragma unroll
    for (int i = KTOP - 1; i > 0; --i) {
      if (K[i] < K[i - 1]) {
        float tk = K[i]; K[i] = K[i - 1]; K[i - 1] = tk;
        float tq = Q[i]; Q[i] = Q[i - 1]; Q[i - 1] = tq;
      }
    }
  }
}

// ---------------------------------------------------------------------------
// Kernel 0: L2-normalize query/target rows, pack to bf16 pairs in workspace.
// ---------------------------------------------------------------------------
__global__ __launch_bounds__(256) void ms_prep(
    const float* __restrict__ q, const float* __restrict__ t,
    unsigned* __restrict__ qp, unsigned* __restrict__ tp) {
  __shared__ float red[256];
  const int r = blockIdx.x, tid = threadIdx.x;

  float a = q[(size_t)r * DIMK + 2 * tid];
  float b = q[(size_t)r * DIMK + 2 * tid + 1];
  red[tid] = a * a + b * b;
  __syncthreads();
  for (int s = 128; s > 0; s >>= 1) { if (tid < s) red[tid] += red[tid + s]; __syncthreads(); }
  float inv = rsqrtf(red[0]);
  qp[(size_t)r * (DIMK / 2) + tid] = pack2_bf16(a * inv, b * inv);
  __syncthreads();

  a = t[(size_t)r * DIMK + 2 * tid];
  b = t[(size_t)r * DIMK + 2 * tid + 1];
  red[tid] = a * a + b * b;
  __syncthreads();
  for (int s = 128; s > 0; s >>= 1) { if (tid < s) red[tid] += red[tid + s]; __syncthreads(); }
  inv = rsqrtf(red[0]);
  tp[(size_t)r * (DIMK / 2) + tid] = pack2_bf16(a * inv, b * inv);
}

// ---------------------------------------------------------------------------
// Kernel 1: streaming bank pass. One workgroup owns GROUP_COLS bank rows,
// processed as NSUB sub-tiles of 16. For each sub-tile:
//   TDM (or VGPR) stage of f32 rows -> row inv-norms + bf16 pairs in LDS
//   8 waves x (2 M-tiles x {q,t}) WMMA bf16, K=512
//   distances staged in LDS; per-thread (=batch row) running top-5.
// Emits 5 (Dk,Dq) candidate pairs per (row, group).
// ---------------------------------------------------------------------------
__global__ __launch_bounds__(256) void ms_bank_pass(
    const float* __restrict__ bank,
    const unsigned* __restrict__ qp, const unsigned* __restrict__ tp,
    float* __restrict__ cand, int ngroups,
    int use_mask,
    const int* __restrict__ labels,
    const int* __restrict__ bank_labels) {
  __shared__ unsigned bankPair[SUB_COLS][DIMK / 2];  // 16 KB bf16 tile
  __shared__ float    partial[SUB_COLS][16];
  __shared__ float    invn[SUB_COLS];
  __shared__ int      slabels[SUB_COLS];
#ifdef MS_USE_TDM
  // 32 KB overlay: f32 TDM staging buffer, dead after bf16 pack, reused as Dk|Dq
  __shared__ __align__(16) float overlay[SUB_COLS * DIMK];
  float* stageF = overlay;
  float (*Dk)[SUB_COLS] = (float (*)[SUB_COLS])overlay;
  float (*Dq)[SUB_COLS] = (float (*)[SUB_COLS])(overlay + B_ROWS * SUB_COLS);
#else
  __shared__ float DkA[B_ROWS][SUB_COLS];            // 16 KB
  __shared__ float DqA[B_ROWS][SUB_COLS];            // 16 KB
  float (*Dk)[SUB_COLS] = DkA;
  float (*Dq)[SUB_COLS] = DqA;
#endif

  const int tid  = threadIdx.x;
  const int grp  = blockIdx.x;
  const int lane = tid & 31;
  const int wv   = tid >> 5;

  const int mylab = labels[tid];

  float K5[KTOP], Q5[KTOP];
#pragma unroll
  for (int i = 0; i < KTOP; ++i) { K5[i] = 3.4e38f; Q5[i] = 0.f; }

  // A-operand rows for this wave (two 16-row M tiles per wave -> 256 rows)
  const int arow0 = wv * 32 + (lane & 15);
  const int arow1 = arow0 + 16;
  const unsigned* q0 = qp + (size_t)arow0 * (DIMK / 2);
  const unsigned* q1 = qp + (size_t)arow1 * (DIMK / 2);
  const unsigned* t0 = tp + (size_t)arow0 * (DIMK / 2);
  const unsigned* t1 = tp + (size_t)arow1 * (DIMK / 2);
  const int khalfA = (lane >> 4) << 3;  // upper half-wave holds K+8 (A layout)
  const int col    = lane & 15;         // B/N column and C/D column

#ifdef MS_USE_TDM
  const unsigned stage_lds_base = (unsigned)(size_t)stageF;  // LDS byte offset
#endif

  for (int s = 0; s < NSUB; ++s) {
    const int cbase = grp * GROUP_COLS + s * SUB_COLS;

    if (s + 1 < NSUB)  // warm L2 for the next sub-tile's rows
      __builtin_prefetch(bank + (size_t)(cbase + SUB_COLS + (tid >> 4)) * DIMK +
                             (tid & 15) * 32, 0, 3);

#ifdef MS_USE_TDM
    // ---- TDM stage: wave wv DMAs bank rows [cbase+2*wv, +2) into LDS ----
    {
      const unsigned lds_dst = stage_lds_base + (unsigned)(wv * 2 * DIMK * 4);
      const unsigned long long ga =
          (unsigned long long)(size_t)(bank + (size_t)(cbase + 2 * wv) * DIMK);
      // D# group0: count=1 | lds_addr | global_addr[56:0] | type=2
      v4u g0 = { 1u,
                 lds_dst,
                 (unsigned)ga,
                 (unsigned)((ga >> 32) & 0x01FFFFFFull) | (2u << 30) };
      // D# group1: data_size=4B; tensor 512 x 2 f32; tile 512 x 2; stride0=512
      v8i g1 = { (int)(2u << 16),
                 (int)((unsigned)DIMK << 16),        // tensor_dim0[15:0]
                 (int)(2u << 16),                    // tensor_dim1[15:0]=2
                 (int)((unsigned)DIMK << 16),        // tile_dim0=512
                 (int)2,                             // tile_dim1=2
                 (int)DIMK,                          // tensor_dim0_stride
                 0, 0 };
      v4i gz = {0, 0, 0, 0};
#if defined(__clang_major__) && __clang_major__ >= 23
      v8i gz8 = {0, 0, 0, 0, 0, 0, 0, 0};
      __builtin_amdgcn_tensor_load_to_lds(g0, g1, gz, gz, gz8, 0);
#else
      __builtin_amdgcn_tensor_load_to_lds(g0, g1, gz, gz, 0);
#endif
      __builtin_amdgcn_s_wait_tensorcnt(0);
    }
    __syncthreads();
    // ---- sumsq + bf16 pack from LDS f32 stage ----
    {
      const int r = tid >> 4, seg = tid & 15;
      const float* src = stageF + r * DIMK + seg * 32;
      float ss = 0.f;
#pragma unroll
      for (int i = 0; i < 32; i += 2) {
        float a = src[i], b = src[i + 1];
        ss += a * a + b * b;
        bankPair[r][seg * 16 + (i >> 1)] = pack2_bf16(a, b);
      }
      partial[r][seg] = ss;
    }
#else
    // ---- VGPR stage: sumsq + bf16 pack straight from global ----
    {
      const int r = tid >> 4, seg = tid & 15;
      const float* src = bank + (size_t)(cbase + r) * DIMK + seg * 32;
      float ss = 0.f;
#pragma unroll
      for (int i = 0; i < 32; i += 2) {
        float a = src[i], b = src[i + 1];
        ss += a * a + b * b;
        bankPair[r][seg * 16 + (i >> 1)] = pack2_bf16(a, b);
      }
      partial[r][seg] = ss;
    }
#endif
    __syncthreads();
    if (tid < SUB_COLS) {
      float acc = 0.f;
#pragma unroll
      for (int i = 0; i < 16; ++i) acc += partial[tid][i];
      invn[tid] = rsqrtf(acc);
      if (use_mask) slabels[tid] = bank_labels[cbase + tid];
    }
    __syncthreads();   // stage reads done; overlay may now be rewritten as Dk/Dq

    // ---- WMMA: (256 batch) x (16 bank cols), K = 512 ----
    v8f accQ0 = zero8(), accT0 = zero8(), accQ1 = zero8(), accT1 = zero8();
    for (int k0 = 0; k0 < DIMK; k0 += 32) {
      FragBF fq0, fq1, ft0, ft1, fb;
#pragma unroll
      for (int v = 0; v < 8; ++v) {
        // A 16x32 bf16 layout: vgpr v holds K pair {16*(v>=4) + khalf + 2*(v&3)}
        const int kkA = ((v >> 2) << 4) + khalfA + ((v & 3) << 1);
        const int pA  = (k0 + kkA) >> 1;
        fq0.u[v] = q0[pA];  fq1.u[v] = q1[pA];
        ft0.u[v] = t0[pA];  ft1.u[v] = t1[pA];
        // B 32x16 bf16 layout: same K order as A (lanes 16-31 hold K+8; vgpr v>=4 holds K+16)
        const int kkB = ((v >> 2) << 4) + ((lane >> 4) << 3) + ((v & 3) << 1);
        fb.u[v] = bankPair[col][(k0 + kkB) >> 1];
      }
      accQ0 = __builtin_amdgcn_wmma_f32_16x16x32_bf16(false, fq0.v, false, fb.v,
                                                      (short)0, accQ0, false, false);
      accT0 = __builtin_amdgcn_wmma_f32_16x16x32_bf16(false, ft0.v, false, fb.v,
                                                      (short)0, accT0, false, false);
      accQ1 = __builtin_amdgcn_wmma_f32_16x16x32_bf16(false, fq1.v, false, fb.v,
                                                      (short)0, accQ1, false, false);
      accT1 = __builtin_amdgcn_wmma_f32_16x16x32_bf16(false, ft1.v, false, fb.v,
                                                      (short)0, accT1, false, false);
    }

    // ---- epilogue: dot -> distance, stage to LDS ----
    // C/D layout: lane holds column col; vgpr j holds M = j + 8*(lane>=16)
    const float inv = invn[col];
    const int mbase = wv * 32 + ((lane >> 4) << 3);
#pragma unroll
    for (int j = 0; j < 8; ++j) {
      Dq[mbase + j][col]      = 2.f - 2.f * accQ0[j] * inv;
      Dk[mbase + j][col]      = 2.f - 2.f * accT0[j] * inv;
      Dq[mbase + 16 + j][col] = 2.f - 2.f * accQ1[j] * inv;
      Dk[mbase + 16 + j][col] = 2.f - 2.f * accT1[j] * inv;
    }
    __syncthreads();

    // ---- per-row running top-5 (thread tid owns batch row tid) ----
#pragma unroll
    for (int c = 0; c < SUB_COLS; ++c) {
      float dk = Dk[tid][c];
      if (use_mask && (mylab != slabels[c])) dk = 5.0f;
      ins5(dk, Dq[tid][c], K5, Q5);
    }
    __syncthreads();
  }

  float* dst = cand + ((size_t)tid * ngroups + grp) * (2 * KTOP);
#pragma unroll
  for (int i = 0; i < KTOP; ++i) { dst[2 * i] = K5[i]; dst[2 * i + 1] = Q5[i]; }
}

// ---------------------------------------------------------------------------
// Kernel 2: per-row candidate merge -> mean of top-5 Dq of the chosen branch.
// ---------------------------------------------------------------------------
__global__ __launch_bounds__(256) void ms_merge(
    const float* __restrict__ cand_u, int ng_u,
    const float* __restrict__ cand_s, int ng_s,
    const int* __restrict__ labels,
    float* __restrict__ contrib) {
  __shared__ float sk[256 * KTOP];
  __shared__ float sq[256 * KTOP];
  const int r = blockIdx.x, tid = threadIdx.x;
  const int lab = labels[r];

  const float* cand; int total;
  if (lab == -1) { cand = cand_u + (size_t)r * ng_u * (2 * KTOP); total = ng_u * KTOP; }
  else           { cand = cand_s + (size_t)r * ng_s * (2 * KTOP); total = ng_s * KTOP; }

  float K5[KTOP], Q5[KTOP];
#pragma unroll
  for (int i = 0; i < KTOP; ++i) { K5[i] = 3.4e38f; Q5[i] = 0.f; }
  for (int i = tid; i < total; i += 256) ins5(cand[2 * i], cand[2 * i + 1], K5, Q5);
#pragma unroll
  for (int i = 0; i < KTOP; ++i) { sk[tid * KTOP + i] = K5[i]; sq[tid * KTOP + i] = Q5[i]; }
  __syncthreads();

  if (tid == 0) {
    float FK[KTOP], FQ[KTOP];
#pragma unroll
    for (int i = 0; i < KTOP; ++i) { FK[i] = 3.4e38f; FQ[i] = 0.f; }
    for (int i = 0; i < 256 * KTOP; ++i) ins5(sk[i], sq[i], FK, FQ);
    contrib[r] = (FQ[0] + FQ[1] + FQ[2] + FQ[3] + FQ[4]) * (1.f / KTOP);
  }
}

// ---------------------------------------------------------------------------
// Kernel 3: L = sum(contrib) / B   (cU + cS == B always)
// ---------------------------------------------------------------------------
__global__ __launch_bounds__(256) void ms_finalize(
    const float* __restrict__ contrib, float* __restrict__ out) {
  __shared__ float red[256];
  const int tid = threadIdx.x;
  red[tid] = contrib[tid];
  __syncthreads();
  for (int s = 128; s > 0; s >>= 1) { if (tid < s) red[tid] += red[tid + s]; __syncthreads(); }
  if (tid == 0) out[0] = red[0] * (1.f / (float)B_ROWS);
}

extern "C" void kernel_launch(void* const* d_in, const int* in_sizes, int n_in,
                              void* d_out, int out_size, void* d_ws, size_t ws_size,
                              hipStream_t stream) {
  (void)n_in; (void)out_size; (void)ws_size;
  const float* query  = (const float*)d_in[0];
  const float* target = (const float*)d_in[1];
  const float* uq     = (const float*)d_in[2];
  const float* sq     = (const float*)d_in[3];
  const int*   labels = (const int*)d_in[4];
  const int*   slab   = (const int*)d_in[5];

  const int Urows = in_sizes[2] / DIMK;      // 128000
  const int Srows = in_sizes[3] / DIMK;      // 12800
  const int ng_u  = Urows / GROUP_COLS;      // 250
  const int ng_s  = Srows / GROUP_COLS;      // 25

  // workspace layout (~3.4 MB total)
  unsigned* qp      = (unsigned*)d_ws;                         // 256 KB
  unsigned* tp      = qp + (size_t)B_ROWS * (DIMK / 2);        // 256 KB
  float*    cand_u  = (float*)(tp + (size_t)B_ROWS * (DIMK / 2));
  float*    cand_s  = cand_u + (size_t)B_ROWS * ng_u * (2 * KTOP);
  float*    contrib = cand_s + (size_t)B_ROWS * ng_s * (2 * KTOP);

  ms_prep<<<B_ROWS, 256, 0, stream>>>(query, target, qp, tp);
  ms_bank_pass<<<ng_u, 256, 0, stream>>>(uq, qp, tp, cand_u, ng_u, 0, labels, slab);
  ms_bank_pass<<<ng_s, 256, 0, stream>>>(sq, qp, tp, cand_s, ng_s, 1, labels, slab);
  ms_merge<<<B_ROWS, 256, 0, stream>>>(cand_u, ng_u, cand_s, ng_s, labels, contrib);
  ms_finalize<<<1, 256, 0, stream>>>(contrib, (float*)d_out);
}
